// Update_12051678233087
// MI455X (gfx1250) — matrix operand order repaired and checked by reading, not compiled, not run on hardware
//
#include <hip/hip_runtime.h>

// ---------------------------------------------------------------------------
// DPVO update operator for MI455X (gfx1250, wave32, WMMA).
// E=98304 edges, D=384. ~530 GFLOP dominated by [E,384]x[384,384] GEMMs ->
// bf16 V_WMMA_F32_16X16X32_BF16 matrix path, fp32 accumulate, fp32 epilogues.
// B (weight) tiles staged global->LDS with GLOBAL_LOAD_ASYNC_TO_LDS_B128
// (ASYNCcnt) -- no VGPR round trip; A tiles converted fp32->bf16 in VGPRs.
// ---------------------------------------------------------------------------

#define EDGES   98304
#define DIMN    384
#define CORRK   882
#define NSEG_KK 6144
#define NSEG_IJ 4096
#define NFRM    64

#define BM  128
#define BN  128
#define BK  32
#define BKP 40          // padded LDS K stride (ushorts); row stride 80B (16B mult)

typedef __attribute__((ext_vector_type(16))) __bf16 v16bf;
typedef __attribute__((ext_vector_type(8)))  float  v8f;

union Frag { unsigned u[8]; v16bf v; };
union Acc  { float    f[8]; v8f   v; };

__device__ __forceinline__ unsigned short f2bf(float f) {
  unsigned u = __float_as_uint(f);
  unsigned r = u + 0x7FFFu + ((u >> 16) & 1u);   // round-to-nearest-even
  return (unsigned short)(r >> 16);
}
__device__ __forceinline__ int f2ord(float f) {
  int i = __float_as_int(f); return i >= 0 ? i : (i ^ 0x7FFFFFFF);
}
__device__ __forceinline__ float ord2f(int o) {
  return __int_as_float(o >= 0 ? o : (o ^ 0x7FFFFFFF));
}
// LDS byte offset of a __shared__ object (AS3 pointers are offsets from LDS base)
__device__ __forceinline__ unsigned lds_off(const void* p) {
  return (unsigned)(unsigned long)(__attribute__((address_space(3))) const void*)p;
}

// ---------------------------------------------------------------------------
// GEMM: Y[M,384] = post( actin(gather(X))[M,K] @ Wb[384,K]^T + bias )
// Wb is bf16 row-major [N=384][K]. Optional epilogue: relu/sigmoid,
// elementwise multiply (resMul), up to two residual adds.
// ---------------------------------------------------------------------------
__global__ __launch_bounds__(256)
void k_gemm_wmma(const float* __restrict__ X, const unsigned short* __restrict__ Wb,
                 const float* __restrict__ bias, float* __restrict__ Y,
                 int K, int actin, int actout,
                 const int*   __restrict__ gatherIdx,
                 const float* __restrict__ resMul,
                 const float* __restrict__ resAdd0,
                 const float* __restrict__ resAdd1)
{
  __shared__ __align__(16) unsigned short As[BM][BKP];
  __shared__ __align__(16) unsigned short Bs[BN][BKP];

  const int tid  = threadIdx.x;
  const int lane = tid & 31;
  const int wave = tid >> 5;
  const int wm   = wave & 3;         // 4 waves along M (32 rows each)
  const int wn   = wave >> 2;        // 2 waves along N (64 cols each)
  const int hl   = lane >> 4;        // lane half (ISA fragment layout)
  const int l16  = lane & 15;

  const int row0 = blockIdx.x * BM;
  const int col0 = blockIdx.y * BN;

  // staging map (A / scalar-B): 2 threads per row, 16 contiguous K elems each
  const int lr = tid >> 1;
  const int lc = (tid & 1) * 16;

  long asrc = (long)(row0 + lr);
  bool avalid = true;
  if (gatherIdx) { int g = gatherIdx[row0 + lr]; avalid = (g >= 0); asrc = avalid ? g : 0; }
  const float*          __restrict__ arow = X  + asrc * (long)K;
  const unsigned short* __restrict__ brow = Wb + (long)(col0 + lr) * K;

  // async-B staging map: 512 x 16B chunks; thread t owns chunks {t, t+256}
  const bool asyncB = ((K & (BK - 1)) == 0) && ((K & 7) == 0);
  const int br0 = tid >> 2;                 // chunk row (first chunk)
  const int bq0 = (tid & 3) * 8;            // ushort offset within row
  const int br1 = (tid + 256) >> 2;
  const int bq1 = ((tid + 256) & 3) * 8;

  Acc acc[2][4];
  #pragma unroll
  for (int i = 0; i < 2; i++)
    #pragma unroll
    for (int j = 0; j < 4; j++)
      #pragma unroll
      for (int q = 0; q < 8; q++) acc[i][j].f[q] = 0.f;

  for (int k0 = 0; k0 < K; k0 += BK) {
    if (k0 + BK <= K) {
      if (asyncB) {                         // B: async DMA global -> LDS (16B)
        const unsigned short* g0 = Wb + (long)(col0 + br0) * K + k0 + bq0;
        const unsigned short* g1 = Wb + (long)(col0 + br1) * K + k0 + bq1;
        unsigned l0 = lds_off(&Bs[br0][bq0]);
        unsigned l1 = lds_off(&Bs[br1][bq1]);
        asm volatile("global_load_async_to_lds_b128 %0, %1, off"
                     :: "v"(l0), "v"(g0) : "memory");
        asm volatile("global_load_async_to_lds_b128 %0, %1, off"
                     :: "v"(l1), "v"(g1) : "memory");
      } else {
        #pragma unroll
        for (int j = 0; j < 16; j++)
          Bs[lr][lc + j] = brow[k0 + lc + j];
      }
      #pragma unroll
      for (int j = 0; j < 8; j++) {         // A: fp32 -> bf16 into LDS
        int k = lc + 2 * j;
        float2 v2 = avalid ? *(const float2*)(arow + k0 + k) : make_float2(0.f, 0.f);
        if (actin) { v2.x = fmaxf(v2.x, 0.f); v2.y = fmaxf(v2.y, 0.f); }
        *(unsigned*)&As[lr][k] = (unsigned)f2bf(v2.x) | ((unsigned)f2bf(v2.y) << 16);
      }
      if (avalid && k0 + BK < K)            // prefetch next A K-step
        __builtin_prefetch(arow + k0 + BK + lc, 0, 1);
    } else {                                // K tail (corr: K=882)
      for (int j = 0; j < 8; j++) {
        int k = lc + 2 * j;
        float x0 = (avalid && (k0 + k     < K)) ? arow[k0 + k]     : 0.f;
        float x1 = (avalid && (k0 + k + 1 < K)) ? arow[k0 + k + 1] : 0.f;
        if (actin) { x0 = fmaxf(x0, 0.f); x1 = fmaxf(x1, 0.f); }
        *(unsigned*)&As[lr][k] = (unsigned)f2bf(x0) | ((unsigned)f2bf(x1) << 16);
      }
      for (int j = 0; j < 16; j++) {
        int k = lc + j;
        Bs[lr][k] = (k0 + k < K) ? brow[k0 + k] : (unsigned short)0;
      }
    }
    if (asyncB) asm volatile("s_wait_asynccnt 0x0" ::: "memory");
    __syncthreads();

    Frag a[2], b[4];
    #pragma unroll
    for (int tm = 0; tm < 2; tm++) {        // A frag: 16-bit A 16x32 layout
      int m = wm * 32 + tm * 16 + l16;
      #pragma unroll
      for (int v = 0; v < 8; v++) {
        int k = (v < 4 ? 2 * v : 16 + 2 * (v - 4)) + hl * 8;
        a[tm].u[v] = *(const unsigned*)&As[m][k];
      }
    }
    #pragma unroll
    for (int tn = 0; tn < 4; tn++) {        // B frag: same K order as A frag
      int n = wn * 64 + tn * 16 + l16;
      #pragma unroll
      for (int v = 0; v < 8; v++) {
        int k = (v < 4 ? 2 * v : 16 + 2 * (v - 4)) + hl * 8;
        b[tn].u[v] = *(const unsigned*)&Bs[n][k];
      }
    }
    #pragma unroll
    for (int tm = 0; tm < 2; tm++)
      #pragma unroll
      for (int tn = 0; tn < 4; tn++)
        acc[tm][tn].v = __builtin_amdgcn_wmma_f32_16x16x32_bf16(
            false, a[tm].v, false, b[tn].v, (short)0, acc[tm][tn].v, false, false);
    __syncthreads();
  }

  #pragma unroll
  for (int tm = 0; tm < 2; tm++) {
    #pragma unroll
    for (int tn = 0; tn < 4; tn++) {
      int col = col0 + wn * 64 + tn * 16 + l16;
      float bv = bias ? bias[col] : 0.f;
      #pragma unroll
      for (int j = 0; j < 8; j++) {         // C/D layout: M = j + hl*8
        int row = row0 + wm * 32 + tm * 16 + hl * 8 + j;
        long idx = (long)row * DIMN + col;
        float v = acc[tm][tn].f[j] + bv;
        if      (actout == 1) v = fmaxf(v, 0.f);
        else if (actout == 2) v = 1.f / (1.f + __expf(-v));
        if (resMul)  v *= resMul[idx];
        if (resAdd0) v += resAdd0[idx];
        if (resAdd1) v += resAdd1[idx];
        Y[idx] = v;
      }
    }
  }
}

// --------------------------- layernorm (eps=1e-3) ---------------------------
__global__ __launch_bounds__(256)
void k_layernorm(const float* __restrict__ X, const float* __restrict__ g,
                 const float* __restrict__ b, float* __restrict__ Y, int rows)
{
  int row = blockIdx.x * 8 + (threadIdx.x >> 5);
  int lane = threadIdx.x & 31;
  if (row >= rows) return;
  const float* xr = X + (long)row * DIMN;
  float vals[12], s = 0.f;
  #pragma unroll
  for (int i = 0; i < 12; i++) { vals[i] = xr[lane + 32 * i]; s += vals[i]; }
  #pragma unroll
  for (int o = 16; o; o >>= 1) s += __shfl_xor(s, o, 32);
  float mean = s * (1.f / DIMN), vs = 0.f;
  #pragma unroll
  for (int i = 0; i < 12; i++) { float d = vals[i] - mean; vs += d * d; }
  #pragma unroll
  for (int o = 16; o; o >>= 1) vs += __shfl_xor(vs, o, 32);
  float rstd = rsqrtf(vs * (1.f / DIMN) + 1e-3f);
  float* yr = Y + (long)row * DIMN;
  #pragma unroll
  for (int i = 0; i < 12; i++) {
    int c = lane + 32 * i;
    yr[c] = (vals[i] - mean) * rstd * g[c] + b[c];
  }
}

// ---------------------- segment softmax aggregation -------------------------
__device__ __forceinline__ int seg_id(const int* id0, const int* id1, int e) {
  return id1 ? (id0[e] * NFRM + id1[e]) : id0[e];
}
__global__ void k_seg_init(int* m, float* den, float* y, long n) {
  long i = (long)blockIdx.x * 256 + threadIdx.x;
  if (i < n) { m[i] = (int)0x80800000; den[i] = 0.f; y[i] = 0.f; }  // ord(-FLT_MAX)
}
__global__ void k_seg_max(const float* __restrict__ gx, const int* id0, const int* id1,
                          int* __restrict__ m) {
  long i = (long)blockIdx.x * 256 + threadIdx.x;
  int e = (int)(i / DIMN), d = (int)(i % DIMN);
  atomicMax(&m[(long)seg_id(id0, id1, e) * DIMN + d], f2ord(gx[i]));
}
__global__ void k_seg_den(const float* __restrict__ gx, const int* id0, const int* id1,
                          const int* __restrict__ m, float* __restrict__ den) {
  long i = (long)blockIdx.x * 256 + threadIdx.x;
  int e = (int)(i / DIMN), d = (int)(i % DIMN);
  long sd = (long)seg_id(id0, id1, e) * DIMN + d;
  atomicAdd(&den[sd], __expf(gx[i] - ord2f(m[sd])));
}
__global__ void k_seg_num(const float* __restrict__ fx, const float* __restrict__ gx,
                          const int* id0, const int* id1, const int* __restrict__ m,
                          const float* __restrict__ den, float* __restrict__ y) {
  long i = (long)blockIdx.x * 256 + threadIdx.x;
  int e = (int)(i / DIMN), d = (int)(i % DIMN);
  long sd = (long)seg_id(id0, id1, e) * DIMN + d;
  float w = __expf(gx[i] - ord2f(m[sd])) / den[sd];
  atomicAdd(&y[sd], fx[i] * w);
}
__global__ void k_expand_add(float* __restrict__ x, const float* __restrict__ h,
                             const int* id0, const int* id1) {
  long i = (long)blockIdx.x * 256 + threadIdx.x;
  int e = (int)(i / DIMN), d = (int)(i % DIMN);
  x[i] += h[(long)seg_id(id0, id1, e) * DIMN + d];
}

// ----------------------------- misc small kernels ---------------------------
__global__ void k_cvt_bf16(const float* __restrict__ src, unsigned short* __restrict__ dst, long n) {
  long i = (long)blockIdx.x * 256 + threadIdx.x;
  if (i < n) dst[i] = f2bf(src[i]);
}
__global__ void k_copy(const float* __restrict__ src, float* __restrict__ dst, long n) {
  long i = (long)blockIdx.x * 256 + threadIdx.x;
  if (i < n) dst[i] = src[i];
}
__global__ __launch_bounds__(256)
void k_head(const float* __restrict__ x, const float* __restrict__ dw, const float* __restrict__ db,
            const float* __restrict__ ww, const float* __restrict__ wb,
            float* __restrict__ delta, float* __restrict__ wout)
{
  int row = blockIdx.x * 8 + (threadIdx.x >> 5);
  int lane = threadIdx.x & 31;
  const float* xr = x + (long)row * DIMN;
  float s0 = 0, s1 = 0, s2 = 0, s3 = 0;
  for (int c = lane; c < DIMN; c += 32) {
    float v = fmaxf(xr[c], 0.f);
    s0 += dw[c] * v; s1 += dw[DIMN + c] * v;
    s2 += ww[c] * v; s3 += ww[DIMN + c] * v;
  }
  for (int o = 16; o; o >>= 1) {
    s0 += __shfl_xor(s0, o, 32); s1 += __shfl_xor(s1, o, 32);
    s2 += __shfl_xor(s2, o, 32); s3 += __shfl_xor(s3, o, 32);
  }
  if (lane == 0) {
    delta[row * 2 + 0] = s0 + db[0];
    delta[row * 2 + 1] = s1 + db[1];
    wout[row * 2 + 0] = 1.f / (1.f + __expf(-(s2 + wb[0])));
    wout[row * 2 + 1] = 1.f / (1.f + __expf(-(s3 + wb[1])));
  }
}

// ---------------------------------------------------------------------------
extern "C" void kernel_launch(void* const* d_in, const int* in_sizes, int n_in,
                              void* d_out, int out_size, void* d_ws, size_t ws_size,
                              hipStream_t stream)
{
  const float* net  = (const float*)d_in[0];
  const float* inp  = (const float*)d_in[1];
  const float* corr = (const float*)d_in[2];
  const int* ii = (const int*)d_in[3];
  const int* jj = (const int*)d_in[4];
  const int* kk = (const int*)d_in[5];
  const int* ix = (const int*)d_in[6];
  const int* jx = (const int*)d_in[7];
  // params flattened depth-first in dict insertion order, leaves w/b (g/b):
  auto P = [&](int i) { return (const float*)d_in[8 + i]; };

  const long E = EDGES, ED = (long)E * DIMN, SEGE = (long)NSEG_KK * DIMN;
  const long WSZ = (long)DIMN * DIMN;

  char* wsb = (char*)d_ws;
  size_t off = 0;
  auto carve = [&](size_t bytes) { char* p = wsb + off; off += (bytes + 255) & ~(size_t)255; return p; };
  float* x   = (float*)carve(ED * 4);
  float* t0  = (float*)carve(ED * 4);
  float* t1  = (float*)carve(ED * 4);
  float* t2  = (float*)carve(ED * 4);
  int*   sgm = (int*)  carve(SEGE * 4);
  float* sgd = (float*)carve(SEGE * 4);
  float* sgy = (float*)carve(SEGE * 4);
  float* sgh = (float*)carve(SEGE * 4);

  // bf16 weight copies: {param leaf index, element count}
  struct WEnt { int leaf; long n; };
  const WEnt wl[19] = {
    {0, (long)DIMN * CORRK}, {2, WSZ}, {6, WSZ},            // corr l0,l1,l2
    {10, WSZ}, {12, WSZ}, {14, WSZ}, {16, WSZ},             // c1, c2
    {18, WSZ}, {20, WSZ}, {22, WSZ},                        // agg_kk f,g,h
    {24, WSZ}, {26, WSZ}, {28, WSZ},                        // agg_ij f,g,h
    {32, WSZ}, {34, WSZ}, {36, WSZ},                        // gru0 gate,r0,r1
    {40, WSZ}, {42, WSZ}, {44, WSZ}                         // gru1 gate,r0,r1
  };
  unsigned short* wb16[19];
  for (int i = 0; i < 19; i++) {
    wb16[i] = (unsigned short*)carve(wl[i].n * 2);
    k_cvt_bf16<<<(unsigned)((wl[i].n + 255) / 256), 256, 0, stream>>>(P(wl[i].leaf), wb16[i], wl[i].n);
  }

  const unsigned edBlk = (unsigned)(ED / 256);
  auto gemm = [&](const float* X, const unsigned short* W, const float* bias, float* Y,
                  long M, int K, int actin, int actout, const int* gather,
                  const float* rmul, const float* radd0, const float* radd1) {
    dim3 grid((unsigned)(M / BM), DIMN / BN);
    k_gemm_wmma<<<grid, 256, 0, stream>>>(X, W, bias, Y, K, actin, actout,
                                          gather, rmul, radd0, radd1);
  };
  auto lnorm = [&](const float* X, const float* g, const float* b, float* Y, long rows) {
    k_layernorm<<<(unsigned)(rows / 8), 256, 0, stream>>>(X, g, b, Y, (int)rows);
  };
  auto softagg = [&](const unsigned short* fw, const float* fb,
                     const unsigned short* gw, const float* gb,
                     const unsigned short* hw, const float* hb,
                     const int* id0, const int* id1, long nseg) {
    gemm(x, fw, fb, t0, E, DIMN, 0, 0, nullptr, nullptr, nullptr, nullptr);   // fx
    gemm(x, gw, gb, t1, E, DIMN, 0, 0, nullptr, nullptr, nullptr, nullptr);   // gx
    k_seg_init<<<(unsigned)((SEGE + 255) / 256), 256, 0, stream>>>(sgm, sgd, sgy, SEGE);
    k_seg_max<<<edBlk, 256, 0, stream>>>(t1, id0, id1, sgm);
    k_seg_den<<<edBlk, 256, 0, stream>>>(t1, id0, id1, sgm, sgd);
    k_seg_num<<<edBlk, 256, 0, stream>>>(t0, t1, id0, id1, sgm, sgd, sgy);
    gemm(sgy, hw, hb, sgh, nseg, DIMN, 0, 0, nullptr, nullptr, nullptr, nullptr);
    k_expand_add<<<edBlk, 256, 0, stream>>>(x, sgh, id0, id1);
  };

  // ---- corr embed: relu(L0) -> L1 -> LN -> relu -> L2, + net + inp ----
  gemm(corr, wb16[0], P(1), t0, E, CORRK, 0, 1, nullptr, nullptr, nullptr, nullptr);
  gemm(t0,   wb16[1], P(3), t1, E, DIMN,  0, 0, nullptr, nullptr, nullptr, nullptr);
  lnorm(t1, P(4), P(5), t1, E);
  gemm(t1,   wb16[2], P(7), x,  E, DIMN,  1, 0, nullptr, nullptr, net, inp);
  lnorm(x, P(8), P(9), x, E);

  // ---- c1 / c2: x += mlp2(mask * x[idx]) ----
  gemm(x,  wb16[3], P(11), t0, E, DIMN, 0, 1, ix, nullptr, nullptr, nullptr);
  gemm(t0, wb16[4], P(13), x,  E, DIMN, 0, 0, nullptr, nullptr, x, nullptr);
  gemm(x,  wb16[5], P(15), t0, E, DIMN, 0, 1, jx, nullptr, nullptr, nullptr);
  gemm(t0, wb16[6], P(17), x,  E, DIMN, 0, 0, nullptr, nullptr, x, nullptr);

  // ---- soft aggregations ----
  softagg(wb16[7],  P(19), wb16[8],  P(21), wb16[9],  P(23), kk, nullptr, NSEG_KK);
  softagg(wb16[10], P(25), wb16[11], P(27), wb16[12], P(29), ii, jj,      NSEG_IJ);

  // ---- GRU: two gated residual blocks ----
  lnorm(x, P(30), P(31), t0, E);
  gemm(t0, wb16[13], P(33), t1, E, DIMN, 0, 2, nullptr, nullptr, nullptr, nullptr); // sigmoid gate
  gemm(t0, wb16[14], P(35), t2, E, DIMN, 0, 1, nullptr, nullptr, nullptr, nullptr); // relu(r0)
  gemm(t2, wb16[15], P(37), x,  E, DIMN, 0, 0, nullptr, t1, x, nullptr);            // x += gate*r1
  lnorm(x, P(38), P(39), t0, E);
  gemm(t0, wb16[16], P(41), t1, E, DIMN, 0, 2, nullptr, nullptr, nullptr, nullptr);
  gemm(t0, wb16[17], P(43), t2, E, DIMN, 0, 1, nullptr, nullptr, nullptr, nullptr);
  gemm(t2, wb16[18], P(45), x,  E, DIMN, 0, 0, nullptr, t1, x, nullptr);

  // ---- outputs: x, delta = relu(x)@d, weight = sigmoid(relu(x)@w) ----
  float* out = (float*)d_out;
  k_copy<<<edBlk, 256, 0, stream>>>(x, out, ED);
  k_head<<<(unsigned)(E / 8), 256, 0, stream>>>(x, P(46), P(47), P(48), P(49),
                                                out + ED, out + ED + 2 * E);
}
